// MLP_learner_17308718202948
// MI455X (gfx1250) — matrix operand order repaired and checked by reading, not compiled, not run on hardware
//
#include <hip/hip_runtime.h>
#include <hip/hip_bf16.h>

// ---------------------------------------------------------------------------
// Problem constants (from reference)
// ---------------------------------------------------------------------------
#define NROWS 50000
#define D     256
#define KP1   21            // K+1 = 21 neighbors kept
#define NEDGE (NROWS * KP1) // 1,050,000
#define NEG_INF (-3.0e38f)

typedef __attribute__((ext_vector_type(16))) _Float16 v16h;
typedef __attribute__((ext_vector_type(8)))  _Float16 v8h;
typedef __attribute__((ext_vector_type(8)))  float    v8f;
typedef __attribute__((ext_vector_type(2)))  float    v2f;

union V16U { v16h v; v8h h[2]; };

// ---------------------------------------------------------------------------
// Kernel 1: MLP layer  out[r][c] = act( sum_k X[r][k]*W[c][k] + bias[c] )
// One wave computes one 16x16 output tile with V_WMMA_F32_16X16X4_F32.
//   A (16x4 f32): lane m=lane&15 holds X[m][k0 + 2*(lane>>4) + {0,1}]  (b64 load)
//   B (4x16 f32): lane n=lane&15 holds W[n][k0 + 2*(lane>>4) + {0,1}]  (b64 load)
//   C (16x16 f32): VGPR v, lane l -> row v + 8*(l>>4), col l&15
// ---------------------------------------------------------------------------
__global__ __launch_bounds__(128) void mlp_layer(const float* __restrict__ X,
                                                 const float* __restrict__ W,
                                                 const float* __restrict__ bias,
                                                 float* __restrict__ out,
                                                 int relu) {
  const int tid  = threadIdx.x;
  const int lane = tid & 31;
  const int wid  = blockIdx.x * 4 + (tid >> 5);   // 0..49999 tiles
  const int rt   = wid >> 4;                      // 0..3124
  const int ctl  = wid & 15;                      // 0..15
  const int rbase = rt * 16;
  const int cbase = ctl * 16;

  const int m  = lane & 15;
  const int kk = (lane >> 4) * 2;
  const float* ap = X + (size_t)(rbase + m) * D + kk;
  const float* bp = W + (size_t)(cbase + m) * D + kk;

  v8f c = {0.f, 0.f, 0.f, 0.f, 0.f, 0.f, 0.f, 0.f};
#pragma unroll 8
  for (int k0 = 0; k0 < D; k0 += 4) {
    v2f a = *(const v2f*)(ap + k0);
    v2f b = *(const v2f*)(bp + k0);
    c = __builtin_amdgcn_wmma_f32_16x16x4_f32(false, a, false, b,
                                              (short)0, c, false, false);
  }

  const float bb = bias[cbase + m];
  const int rr = (lane >> 4) * 8;
#pragma unroll
  for (int v = 0; v < 8; ++v) {
    float val = c[v] + bb;
    if (relu) val = fmaxf(val, 0.0f);
    out[(size_t)(rbase + rr + v) * D + (cbase + m)] = val;
  }
}

// ---------------------------------------------------------------------------
// Kernel 2: row L2-normalize embeddings; emit f16 Xn (row-major) and XnT
// (transposed; unread, WMMA B-operand loads come from Xn rows).
// ---------------------------------------------------------------------------
__global__ __launch_bounds__(256) void normalize_k(const float* __restrict__ emb,
                                                   _Float16* __restrict__ Xn,
                                                   _Float16* __restrict__ XnT) {
  __shared__ float red[256];
  const int r = blockIdx.x;
  const int t = threadIdx.x;
  float x = emb[(size_t)r * D + t];
  red[t] = x * x;
  __syncthreads();
  for (int s = 128; s > 0; s >>= 1) {
    if (t < s) red[t] += red[t + s];
    __syncthreads();
  }
  float nrm = fmaxf(sqrtf(red[0]), 1e-12f);
  float xn = x / nrm;
  Xn [(size_t)r * D + t]      = (_Float16)xn;
  XnT[(size_t)t * NROWS + r]  = (_Float16)xn;
}

// ---------------------------------------------------------------------------
// Kernel 3: fused similarity GEMM + top-21.
// Block = 256 threads (8 waves), owns a 16-row stripe. Per iteration the 8
// waves compute 8 adjacent 16x16 tiles (128 columns) of sims = Xn @ Xn^T with
// V_WMMA_F32_16X16X32_F16 (K=256 -> 8 wmma per tile), stage them in LDS, then
// all 256 threads (16 per row) stream candidates into LDS-resident top-21
// lists guarded by a register running-min. Final per-row merge of 16 lists.
// A fragments (the 16 rows) persist in VGPRs for the whole column loop.
// ---------------------------------------------------------------------------
__global__ __launch_bounds__(256) void sims_topk(const _Float16* __restrict__ Xn,
                                                 const _Float16* __restrict__ XnT,
                                                 float* __restrict__ vals,
                                                 int* __restrict__ inds) {
  __shared__ float simsf[16 * 128];
  __shared__ float lval[256 * KP1];
  __shared__ int   lind[256 * KP1];

  const int tid   = threadIdx.x;
  const int lane  = tid & 31;
  const int wave  = tid >> 5;           // col-tile slot 0..7 within iteration
  const int rbase = blockIdx.x * 16;

  float* myv = &lval[tid * KP1];
  int*   myi = &lind[tid * KP1];
#pragma unroll
  for (int i = 0; i < KP1; ++i) { myv[i] = NEG_INF; myi[i] = 0; }
  float vmin = NEG_INF;
  int   vminpos = 0;

  // Persistent A fragments: 8 K-chunks of 32 (16-bit A 16x32 layout).
  V16U afrag[8];
  {
    const int m = lane & 15;
    const int h = lane >> 4;
    const _Float16* rp = Xn + (size_t)(rbase + m) * D;
#pragma unroll
    for (int kc = 0; kc < 8; ++kc) {
      const _Float16* p = rp + kc * 32;
      afrag[kc].h[0] = *(const v8h*)(p + h * 8);        // K 0-7 / 8-15
      afrag[kc].h[1] = *(const v8h*)(p + 16 + h * 8);   // K 16-23 / 24-31
    }
  }

  const int row = tid >> 4;   // selection row 0..15
  const int sub = tid & 15;

  const int NCT = NROWS / 16; // 3125 column tiles
  for (int ct0 = 0; ct0 < NCT; ct0 += 8) {
    // ---- compute phase: this wave's 16x16 tile (clamped, branch-free so
    //      EXEC stays all-ones around WMMA; clamped tiles are never read) ----
    int myct = ct0 + wave;
    myct = (myct < NCT) ? myct : (NCT - 1);
    const int cbase = myct * 16;

    v8f c = {0.f, 0.f, 0.f, 0.f, 0.f, 0.f, 0.f, 0.f};
#pragma unroll
    for (int kc = 0; kc < 8; ++kc) {
      // B 32x16: lane -> N = lane&15; K 0-7/8-15 then 16-23/24-31 (same as A).
      V16U b;
      const _Float16* p = Xn + (size_t)(cbase + (lane & 15)) * D + kc * 32;
      b.h[0] = *(const v8h*)(p + (lane >> 4) * 8);
      b.h[1] = *(const v8h*)(p + 16 + (lane >> 4) * 8);
      c = __builtin_amdgcn_wmma_f32_16x16x32_f16(false, afrag[kc].v, false, b.v,
                                                 (short)0, c, false, false);
    }
    {
      const int cc = wave * 16 + (lane & 15);
      const int rr = (lane >> 4) * 8;
#pragma unroll
      for (int v = 0; v < 8; ++v)
        simsf[(rr + v) * 128 + cc] = c[v];
    }
    __syncthreads();

    // ---- selection phase: 16 threads per row scan 8 candidates each ----
    const int colbase = ct0 * 16;
    const int ncols = (NROWS - colbase < 128) ? (NROWS - colbase) : 128;
#pragma unroll
    for (int j = 0; j < 8; ++j) {
      const int col = sub + 16 * j;
      if (col < ncols) {
        const float v = simsf[row * 128 + col];
        if (v > vmin) {
          myv[vminpos] = v;
          myi[vminpos] = colbase + col;
          float mn = myv[0]; int mp = 0;
#pragma unroll
          for (int i = 1; i < KP1; ++i) {
            const float t = myv[i];
            if (t < mn) { mn = t; mp = i; }
          }
          vmin = mn; vminpos = mp;
        }
      }
    }
    __syncthreads();
  }

  // ---- merge: thread t<16 owns row t; top-21 of 16*21=336 candidates,
  //      descending, ties -> smaller index (matches top_k ordering) ----
  if (tid < 16) {
    const int base = tid * 16 * KP1;
    for (int j = 0; j < KP1; ++j) {
      float bv = NEG_INF; int bi = 0x7fffffff; int bp = 0;
      for (int i = 0; i < 16 * KP1; ++i) {
        const float v = lval[base + i];
        const int  ix = lind[base + i];
        if (v > bv || (v == bv && ix < bi)) { bv = v; bi = ix; bp = i; }
      }
      lval[base + bp] = NEG_INF;
      vals[(size_t)(rbase + tid) * KP1 + j] = bv;
      inds[(size_t)(rbase + tid) * KP1 + j] = bi;
    }
  }
}

// ---------------------------------------------------------------------------
// Kernels 4a/4b/5: degree normalization + edge emission.
// ---------------------------------------------------------------------------
__global__ void norm_row_k(const float* __restrict__ vals, float* __restrict__ norm) {
  const int r = blockIdx.x * blockDim.x + threadIdx.x;
  if (r < NROWS) {
    float s = 0.f;
#pragma unroll
    for (int j = 0; j < KP1; ++j) s += vals[(size_t)r * KP1 + j];
    norm[r] = s;
  }
}

__global__ void norm_col_k(const float* __restrict__ vals, const int* __restrict__ inds,
                           float* __restrict__ norm) {
  const int e = blockIdx.x * blockDim.x + threadIdx.x;
  if (e < NEDGE) atomicAdd(&norm[inds[e]], vals[e]);
}

__global__ void edges_k(const float* __restrict__ vals, const int* __restrict__ inds,
                        const float* __restrict__ norm,
                        int* __restrict__ ei, float* __restrict__ ew) {
  const int e = blockIdx.x * blockDim.x + threadIdx.x;
  if (e < NEDGE) {
    const int r = e / KP1;
    const int c = inds[e];
    const float w = vals[e] * (1.0f / sqrtf(norm[r])) * (1.0f / sqrtf(norm[c]));
    ei[e] = r;
    ei[NEDGE + e] = c;
    ew[e] = w;
  }
}

// ---------------------------------------------------------------------------
// Host launcher.
// d_out layout (reference return order, flat): edge_index [2*NEDGE] i32,
// edge_weight [NEDGE] f32, embeddings [NROWS*D] f32.
// Workspace layout (h region reused by Xn once dead): 
//   [0,51.2M)   h (f32 hidden)   ->  [0,25.6M) Xn f16 after layer2
//   [25.6M,51.2M) XnT f16
//   [51.2M,55.4M) vals f32 ; [55.4M,59.6M) inds i32 ; [59.6M,59.8M) norm f32
// ---------------------------------------------------------------------------
extern "C" void kernel_launch(void* const* d_in, const int* in_sizes, int n_in,
                              void* d_out, int out_size, void* d_ws, size_t ws_size,
                              hipStream_t stream) {
  const float* features = (const float*)d_in[0];
  const float* W0 = (const float*)d_in[1];
  const float* b0 = (const float*)d_in[2];
  const float* W1 = (const float*)d_in[3];
  const float* b1 = (const float*)d_in[4];

  char* ws = (char*)d_ws;
  float*    h    = (float*)ws;                         // 50000*256 f32
  _Float16* Xn   = (_Float16*)ws;                      // reuses h region
  _Float16* XnT  = (_Float16*)(ws + 25600000);
  float*    vals = (float*)(ws + 51200000);
  int*      inds = (int*)  (ws + 55400000);
  float*    norm = (float*)(ws + 59600000);

  int*   ei  = (int*)d_out;
  float* ew  = (float*)d_out + 2 * NEDGE;
  float* emb = (float*)d_out + 3 * NEDGE;

  // MLP: layer0 (ReLU) -> layer1, embeddings land directly in d_out.
  mlp_layer<<<12500, 128, 0, stream>>>(features, W0, b0, h, 1);
  mlp_layer<<<12500, 128, 0, stream>>>(h, W1, b1, emb, 0);

  // Normalize + f16 conversion (+ transposed copy XnT, unused by sims_topk).
  normalize_k<<<NROWS, 256, 0, stream>>>(emb, Xn, XnT);

  // Fused Gram-matrix + top-21.
  sims_topk<<<NROWS / 16, 256, 0, stream>>>(Xn, XnT, vals, inds);

  // Degree norms and edge outputs.
  norm_row_k<<<(NROWS + 255) / 256, 256, 0, stream>>>(vals, norm);
  norm_col_k<<<(NEDGE + 255) / 256, 256, 0, stream>>>(vals, inds, norm);
  edges_k  <<<(NEDGE + 255) / 256, 256, 0, stream>>>(vals, inds, norm, ei, ew);
}
